// Deepnet_34102040330541
// MI455X (gfx1250) — matrix operand order repaired and checked by reading, not compiled, not run on hardware
//
#include <hip/hip_runtime.h>

typedef __attribute__((ext_vector_type(16))) __bf16 v16bf;
typedef __attribute__((ext_vector_type(8)))  __bf16 v8bf;
typedef __attribute__((ext_vector_type(8)))  float  v8f;
typedef __attribute__((ext_vector_type(4)))  int    v4i;

#define BATCH   4096
#define SEQ     41
#define FDIM    128
#define EPS_BN  1e-5f
#define EPS_SQ  1e-21f

#if defined(__has_builtin)
#if __has_builtin(__builtin_amdgcn_global_load_async_to_lds_b128) && \
    __has_builtin(__builtin_amdgcn_s_wait_asynccnt)
#define HAVE_ASYNC_LDS 1
#endif
#endif

#ifdef HAVE_ASYNC_LDS
typedef __attribute__((address_space(1))) v4i gm_v4i;   // global int4
typedef __attribute__((address_space(3))) v4i lds_v4i;  // LDS int4
#endif

// stats region layout (floats): [0..63] bn sum, [64..127] bn sumsq,
// [128..129] bn1 sum, [130..131] bn1 sumsq, [132..133] bn2 sum, [134..135] bn2 sumsq
__global__ void zero_stats_kernel(float* __restrict__ s) {
    s[threadIdx.x] = 0.0f;
}

// Convert f32 weight [S, K, N] into WMMA B-fragment order:
// element offset = (((s*KC + kc)*NT + nt)*32 + lane)*16 ; per-lane 16 bf16 with
// N = nt*16 + lane%16, K = kc*32 + 8*(lane/16) + e (e<8), kc*32 + 16 + 8*(lane/16) + (e-8) (e>=8)
__global__ void pack_b_kernel(const float* __restrict__ W, __bf16* __restrict__ out,
                              int S, int K, int N) {
    int KC = K >> 5, NT = N >> 4;
    size_t total = (size_t)S * KC * NT * 32;
    size_t id = (size_t)blockIdx.x * blockDim.x + threadIdx.x;
    if (id >= total) return;
    int lane = (int)(id & 31);
    size_t t = id >> 5;
    int nt = (int)(t % NT); t /= NT;
    int kc = (int)(t % KC); int s = (int)(t / KC);
    int n  = nt * 16 + (lane & 15);
    int kb = kc * 32 + 8 * (lane >> 4);
    __bf16* o = out + id * 16;
    #pragma unroll
    for (int e = 0; e < 16; ++e)
        o[e] = (__bf16)W[((size_t)s * K + (kb + e + (e >= 8 ? 8 : 0))) * N + n];
}

__global__ void embed_kernel(const int* __restrict__ x, const float* __restrict__ emb,
                             __bf16* __restrict__ out, int total) {
    int id = blockIdx.x * blockDim.x + threadIdx.x;
    if (id >= total) return;
    int f  = id & 127;
    int bs = id >> 7;                 // b*41 + s
    out[id] = (__bf16)emb[x[bs] * FDIM + f];
}

// Per-position GEMM: out[b,s,:] = relu( window(act)[b,s,:] @ W[s] + bias[s,:] )
// One wave = 32 batch rows x 128 filters (2 A fragments, 16 accumulators).
// Each LDS B-fragment read feeds two WMMAs -> 16 B/lane per matrix op.
// Weights staged block-wide through double-buffered LDS (async path if available).
__global__ __launch_bounds__(256)
void poslin_wmma(const __bf16* __restrict__ actin, const __bf16* __restrict__ Wp,
                 const float* __restrict__ bias, __bf16* __restrict__ actout,
                 int KC, int pad) {
    __shared__ __bf16 bstage[2][8 * 32 * 16];   // 2 x 8KB ping-pong
    const int t    = threadIdx.x;
    const int lane = t & 31;
    const int wave = t >> 5;
    const int s    = blockIdx.y;
    const int b0   = (blockIdx.x * 8 + wave) * 32;
    const int mrow = lane & 15;
    const int grp  = lane >> 4;

    const __bf16* bbase = Wp + (size_t)s * KC * 4096;   // 8 ntiles * 32 lanes * 16

    // cooperative stage of one 4096-elem (8KB) weight chunk: 16 bf16 (32B) per thread
    auto stage = [&](int kc, int buf) {
        const __bf16* src = bbase + (size_t)kc * 4096 + (size_t)t * 16;
        __bf16* dst = &bstage[buf][t * 16];
#ifdef HAVE_ASYNC_LDS
        __builtin_amdgcn_global_load_async_to_lds_b128((gm_v4i*)src,       (lds_v4i*)dst,       0, 0);
        __builtin_amdgcn_global_load_async_to_lds_b128((gm_v4i*)(src + 8), (lds_v4i*)(dst + 8), 0, 0);
#else
        v8bf x0 = *(const v8bf*)(src);
        v8bf x1 = *(const v8bf*)(src + 8);
        *(v8bf*)(dst)     = x0;
        *(v8bf*)(dst + 8) = x1;
#endif
    };

    v8f acc[2][8];
    #pragma unroll
    for (int mt = 0; mt < 2; ++mt)
        #pragma unroll
        for (int i = 0; i < 8; ++i)
            #pragma unroll
            for (int r = 0; r < 8; ++r) acc[mt][i][r] = 0.0f;

    const __bf16* arow0 = actin + (size_t)(b0 + mrow) * SEQ * FDIM;
    const __bf16* arow1 = actin + (size_t)(b0 + 16 + mrow) * SEQ * FDIM;

    stage(0, 0);
#ifdef HAVE_ASYNC_LDS
    __builtin_amdgcn_s_wait_asynccnt(0);
#endif
    __syncthreads();

    for (int kc = 0; kc < KC; ++kc) {
        const int buf = kc & 1;
        if (kc + 1 < KC) stage(kc + 1, buf ^ 1);   // issue next-chunk staging (overlaps WMMA)

        // A fragments (16-bit A layout): elems 0..7 -> K=8*grp+e, 8..15 -> K=16+8*grp+e
        int s_src = s - pad + (kc >> 2);
        int f0    = (kc & 3) * 32 + 8 * grp;
        union { v16bf v; v8bf h[2]; } a0, a1;
        if ((unsigned)s_src < (unsigned)SEQ) {
            const __bf16* ap0 = arow0 + (size_t)s_src * FDIM + f0;
            const __bf16* ap1 = arow1 + (size_t)s_src * FDIM + f0;
            a0.h[0] = *(const v8bf*)(ap0);
            a0.h[1] = *(const v8bf*)(ap0 + 16);
            a1.h[0] = *(const v8bf*)(ap1);
            a1.h[1] = *(const v8bf*)(ap1 + 16);
        } else {
            #pragma unroll
            for (int e = 0; e < 16; ++e) { a0.v[e] = (__bf16)0.0f; a1.v[e] = (__bf16)0.0f; }
        }

        // each B fragment read from LDS feeds two WMMAs (rows 0..15 and 16..31)
        const __bf16* lbp = &bstage[buf][lane * 16];
        #pragma unroll
        for (int nt = 0; nt < 8; ++nt) {
            v16bf bv = *(const v16bf*)(lbp + (size_t)nt * 512);
            acc[0][nt] = __builtin_amdgcn_wmma_f32_16x16x32_bf16(
                false, a0.v, false, bv, (short)0, acc[0][nt], false, false);
            acc[1][nt] = __builtin_amdgcn_wmma_f32_16x16x32_bf16(
                false, a1.v, false, bv, (short)0, acc[1][nt], false, false);
        }

#ifdef HAVE_ASYNC_LDS
        __builtin_amdgcn_s_wait_asynccnt(0);
#endif
        __syncthreads();   // next-chunk staging complete; reads of this buffer done
    }

    const float* brow = bias + s * FDIM;
    #pragma unroll
    for (int mt = 0; mt < 2; ++mt) {
        #pragma unroll
        for (int nt = 0; nt < 8; ++nt) {
            int n  = nt * 16 + mrow;
            float bn = brow[n];
            #pragma unroll
            for (int r = 0; r < 8; ++r) {
                int M   = mt * 16 + r + 8 * grp;
                float v = acc[mt][nt][r] + bn;
                v = v > 0.0f ? v : 0.0f;
                actout[((size_t)(b0 + M) * SEQ + s) * FDIM + n] = (__bf16)v;
            }
        }
    }
}

// conv1d over C=41, k=3 (NCH), + bias, bf16 out, bn stats via atomics.
__global__ __launch_bounds__(128)
void conv1_kernel(const __bf16* __restrict__ act, const float* __restrict__ w,
                  const float* __restrict__ cb, __bf16* __restrict__ out,
                  float* __restrict__ stats) {
    __shared__ float sin[SEQ * FDIM];
    __shared__ float sw[64 * SEQ * 3];
    int t = threadIdx.x, b = blockIdx.x;
    for (int i = t; i < SEQ * FDIM; i += 128) sin[i] = (float)act[(size_t)b * SEQ * FDIM + i];
    for (int i = t; i < 64 * SEQ * 3; i += 128) sw[i] = w[i];
    __syncthreads();
    int l = t, lane = t & 31;
    for (int o = 0; o < 64; ++o) {
        float acc = cb[o];
        const float* wr = sw + o * SEQ * 3;
        #pragma unroll 1
        for (int c = 0; c < SEQ; ++c) {
            float x0 = (l > 0)   ? sin[c * FDIM + l - 1] : 0.0f;
            float x1 =             sin[c * FDIM + l];
            float x2 = (l < 127) ? sin[c * FDIM + l + 1] : 0.0f;
            acc += x0 * wr[c * 3] + x1 * wr[c * 3 + 1] + x2 * wr[c * 3 + 2];
        }
        out[((size_t)b * 64 + o) * FDIM + l] = (__bf16)acc;
        float s1 = acc, s2 = acc * acc;
        for (int off = 16; off; off >>= 1) {
            s1 += __shfl_down(s1, off, 32);
            s2 += __shfl_down(s2, off, 32);
        }
        if (lane == 0) { atomicAdd(&stats[o], s1); atomicAdd(&stats[64 + o], s2); }
    }
}

// bn+relu -> caps conv (groups=2) -> squash -> routing -> squash -> pair/max -> ybuf (bf16)
__global__ __launch_bounds__(128)
void caps_route_kernel(const __bf16* __restrict__ conv1out, const float* __restrict__ stats,
                       const float* __restrict__ bng, const float* __restrict__ bnb,
                       const float* __restrict__ capsw, const float* __restrict__ capsb,
                       const float* __restrict__ routeW, const float* __restrict__ routeb,
                       __bf16* __restrict__ ybuf) {
    __shared__ float h1[64 * FDIM];
    __shared__ float cw[64 * 32 * 3];
    __shared__ float sscale[64], sshift[64];
    __shared__ float red[4];
    __shared__ float bcast;
    int t = threadIdx.x, b = blockIdx.x;
    const float invN = 1.0f / ((float)BATCH * (float)FDIM);
    if (t < 64) {
        float mu  = stats[t] * invN;
        float var = stats[64 + t] * invN - mu * mu;
        float sc  = bng[t] * rsqrtf(var + EPS_BN);
        sscale[t] = sc; sshift[t] = bnb[t] - mu * sc;
    }
    for (int i = t; i < 64 * 32 * 3; i += 128) cw[i] = capsw[i];
    __syncthreads();
    for (int i = t; i < 64 * FDIM; i += 128) {
        int o = i >> 7;
        float v = (float)conv1out[(size_t)b * 64 * FDIM + i] * sscale[o] + sshift[o];
        h1[i] = v > 0.0f ? v : 0.0f;
    }
    __syncthreads();
    int l = t, lane = t & 31, wid = t >> 5;
    float s0 = 0.0f, s1 = 0.0f;
    for (int o = 0; o < 64; ++o) {
        int cbase = o & 32;                  // group of 32 input channels
        float acc = capsb[o];
        const float* wr = cw + o * 96;
        #pragma unroll 1
        for (int ci = 0; ci < 32; ++ci) {
            const float* hr = h1 + (cbase + ci) * FDIM;
            float x0 = (l > 0)   ? hr[l - 1] : 0.0f;
            float x1 =             hr[l];
            float x2 = (l < 127) ? hr[l + 1] : 0.0f;
            acc += x0 * wr[ci * 3] + x1 * wr[ci * 3 + 1] + x2 * wr[ci * 3 + 2];
        }
        float r = acc * acc;                 // row norm^2 for squash
        for (int off = 16; off; off >>= 1) r += __shfl_down(r, off, 32);
        if (lane == 0) red[wid] = r;
        __syncthreads();
        if (t == 0) bcast = red[0] + red[1] + red[2] + red[3];
        __syncthreads();
        float nrm = sqrtf(bcast);
        float fac = (1.0f - 1.0f / (expf(nrm) + EPS_SQ)) / (nrm + EPS_SQ);
        float hs  = fac * acc;
        float c0  = 1.0f + routeb[o];
        float c1  = 1.0f + routeb[64 + o];
        s0 += hs * routeW[(size_t)o * FDIM + l] * c0;
        s1 += hs * routeW[(size_t)(64 + o) * FDIM + l] * c1;
        __syncthreads();
    }
    float sv[2] = { s0, s1 }, facs[2];
    for (int j = 0; j < 2; ++j) {
        float r = sv[j] * sv[j];
        for (int off = 16; off; off >>= 1) r += __shfl_down(r, off, 32);
        if (lane == 0) red[wid] = r;
        __syncthreads();
        if (t == 0) bcast = red[0] + red[1] + red[2] + red[3];
        __syncthreads();
        float nrm = sqrtf(bcast);
        facs[j] = (1.0f - 1.0f / (expf(nrm) + EPS_SQ)) / (nrm + EPS_SQ);
        __syncthreads();
    }
    float ss0 = facs[0] * s0, ss1 = facs[1] * s1;
    float m = fmaxf(ss0, ss1);
    ybuf[((size_t)b * 2 + 0) * FDIM + l] = (__bf16)(ss0 + m);
    ybuf[((size_t)b * 2 + 1) * FDIM + l] = (__bf16)(ss1 + m);
}

// fc1: [8192,128] bf16 @ packed [128,64] -> z1 f32, bn1 stats per parity(row)
__global__ __launch_bounds__(256)
void fc1_wmma(const __bf16* __restrict__ ybuf, const __bf16* __restrict__ fc1p,
              float* __restrict__ z1, float* __restrict__ stats) {
    int lane = threadIdx.x & 31, wave = threadIdx.x >> 5;
    int row0 = (blockIdx.x * 8 + wave) * 16;
    int mrow = lane & 15, grp = lane >> 4;
    v8f acc[4];
    #pragma unroll
    for (int i = 0; i < 4; ++i)
        #pragma unroll
        for (int r = 0; r < 8; ++r) acc[i][r] = 0.0f;
    const __bf16* arow = ybuf + (size_t)(row0 + mrow) * FDIM;
    for (int kc = 0; kc < 4; ++kc) {
        union { v16bf v; v8bf h[2]; } a;
        const __bf16* ap = arow + kc * 32 + 8 * grp;
        a.h[0] = *(const v8bf*)(ap);
        a.h[1] = *(const v8bf*)(ap + 16);
        const __bf16* bp = fc1p + ((size_t)kc * 128 + lane) * 16;
        v16bf bv[4];
        #pragma unroll
        for (int nt = 0; nt < 4; ++nt)
            bv[nt] = *(const v16bf*)(bp + (size_t)nt * 512);
        #pragma unroll
        for (int nt = 0; nt < 4; ++nt)
            acc[nt] = __builtin_amdgcn_wmma_f32_16x16x32_bf16(
                false, a.v, false, bv[nt], (short)0, acc[nt], false, false);
    }
    float ls[2] = { 0, 0 }, lq[2] = { 0, 0 };
    #pragma unroll
    for (int nt = 0; nt < 4; ++nt) {
        int n = nt * 16 + mrow;
        #pragma unroll
        for (int r = 0; r < 8; ++r) {
            int M = r + 8 * grp;
            float v = acc[nt][r];
            z1[(size_t)(row0 + M) * 64 + n] = v;
            ls[r & 1] += v; lq[r & 1] += v * v;
        }
    }
    for (int j = 0; j < 2; ++j) {
        float a1 = ls[j], a2 = lq[j];
        for (int off = 16; off; off >>= 1) {
            a1 += __shfl_down(a1, off, 32);
            a2 += __shfl_down(a2, off, 32);
        }
        if (lane == 0) { atomicAdd(&stats[128 + j], a1); atomicAdd(&stats[130 + j], a2); }
    }
}

// bn1+relu -> fc2 (64->32) -> z2 f32 + bn2 stats
__global__ __launch_bounds__(256)
void fc2_kernel(const float* __restrict__ z1, float* __restrict__ stats,
                const float* __restrict__ g1, const float* __restrict__ b1,
                const float* __restrict__ fc2w, float* __restrict__ z2) {
    int t = threadIdx.x;
    int m = t & 31, rl = t >> 5;
    int row = blockIdx.x * 8 + rl;
    int j = row & 1;
    const float invN = 1.0f / ((float)BATCH * 64.0f);
    float mu  = stats[128 + j] * invN;
    float var = stats[130 + j] * invN - mu * mu;
    float sc  = g1[j] * rsqrtf(var + EPS_BN);
    float sh  = b1[j] - mu * sc;
    float acc = 0.0f;
    const float* zr = z1 + (size_t)row * 64;
    #pragma unroll 1
    for (int k = 0; k < 64; ++k) {
        float v = zr[k] * sc + sh;
        v = v > 0.0f ? v : 0.0f;
        acc += v * fc2w[k * 32 + m];
    }
    z2[(size_t)row * 32 + m] = acc;
    float a1 = acc, a2 = acc * acc;
    for (int off = 16; off; off >>= 1) {
        a1 += __shfl_down(a1, off, 32);
        a2 += __shfl_down(a2, off, 32);
    }
    if (m == 0) { atomicAdd(&stats[132 + j], a1); atomicAdd(&stats[134 + j], a2); }
}

// bn2+relu -> fc3 dot -> sigmoid -> max over j
__global__ __launch_bounds__(256)
void head_kernel(const float* __restrict__ z2, const float* __restrict__ stats,
                 const float* __restrict__ g2, const float* __restrict__ b2,
                 const float* __restrict__ fc3w, float* __restrict__ out) {
    __shared__ float sig[8];
    int t = threadIdx.x, lane = t & 31, wid = t >> 5;
    int bl = wid >> 1, j = wid & 1;
    int b = blockIdx.x * 4 + bl;
    int row = b * 2 + j;
    const float invN = 1.0f / ((float)BATCH * 32.0f);
    float mu  = stats[132 + j] * invN;
    float var = stats[134 + j] * invN - mu * mu;
    float sc  = g2[j] * rsqrtf(var + EPS_BN);
    float sh  = b2[j] - mu * sc;
    float v = z2[(size_t)row * 32 + lane] * sc + sh;
    v = v > 0.0f ? v : 0.0f;
    v *= fc3w[lane];
    for (int off = 16; off; off >>= 1) v += __shfl_down(v, off, 32);
    if (lane == 0) sig[wid] = 1.0f / (1.0f + expf(-v));
    __syncthreads();
    if (t < 4) out[blockIdx.x * 4 + t] = fmaxf(sig[2 * t], sig[2 * t + 1]);
}

extern "C" void kernel_launch(void* const* d_in, const int* in_sizes, int n_in,
                              void* d_out, int out_size, void* d_ws, size_t ws_size,
                              hipStream_t stream) {
    const int*   x       = (const int*)d_in[0];
    const float* emb     = (const float*)d_in[1];
    const float* W3      = (const float*)d_in[2];
    const float* b3      = (const float*)d_in[3];
    const float* W5      = (const float*)d_in[4];
    const float* b5      = (const float*)d_in[5];
    const float* W7      = (const float*)d_in[6];
    const float* b7      = (const float*)d_in[7];
    const float* conv1_w = (const float*)d_in[8];
    const float* conv1_b = (const float*)d_in[9];
    const float* bn_g    = (const float*)d_in[10];
    const float* bn_b    = (const float*)d_in[11];
    const float* caps_w  = (const float*)d_in[12];
    const float* caps_b  = (const float*)d_in[13];
    const float* route_W = (const float*)d_in[14];
    const float* route_b = (const float*)d_in[15];
    const float* fc1_w   = (const float*)d_in[16];
    const float* bn1_g   = (const float*)d_in[17];
    const float* bn1_b   = (const float*)d_in[18];
    const float* fc2_w   = (const float*)d_in[19];
    const float* bn2_g   = (const float*)d_in[20];
    const float* bn2_b   = (const float*)d_in[21];
    const float* fc3_w   = (const float*)d_in[22];
    float* outp = (float*)d_out;

    char* ws = (char*)d_ws;
    size_t off = 0;
    float* stats = (float*)(ws + off); off += 1024;
    __bf16* actA = (__bf16*)(ws + off); off += (size_t)BATCH * SEQ * FDIM * 2;
    __bf16* actB = (__bf16*)(ws + off); off += (size_t)BATCH * SEQ * FDIM * 2;
    __bf16* convout = (__bf16*)(ws + off); off += (size_t)BATCH * 64 * FDIM * 2;
    __bf16* ybuf = (__bf16*)(ws + off); off += (size_t)BATCH * 2 * FDIM * 2;
    float*  z1   = (float*)(ws + off);  off += (size_t)BATCH * 2 * 64 * 4;
    float*  z2   = (float*)(ws + off);  off += (size_t)BATCH * 2 * 32 * 4;
    __bf16* W3p  = (__bf16*)(ws + off); off += (size_t)SEQ * 384 * FDIM * 2;
    __bf16* W5p  = (__bf16*)(ws + off); off += (size_t)SEQ * 640 * FDIM * 2;
    __bf16* W7p  = (__bf16*)(ws + off); off += (size_t)SEQ * 896 * FDIM * 2;
    __bf16* fc1p = (__bf16*)(ws + off); off += (size_t)FDIM * 64 * 2;

    zero_stats_kernel<<<1, 256, 0, stream>>>(stats);

    auto packGrid = [](size_t tot) { return (unsigned)((tot + 255) / 256); };
    pack_b_kernel<<<packGrid((size_t)SEQ * 12 * 8 * 32), 256, 0, stream>>>(W3, W3p, SEQ, 384, FDIM);
    pack_b_kernel<<<packGrid((size_t)SEQ * 20 * 8 * 32), 256, 0, stream>>>(W5, W5p, SEQ, 640, FDIM);
    pack_b_kernel<<<packGrid((size_t)SEQ * 28 * 8 * 32), 256, 0, stream>>>(W7, W7p, SEQ, 896, FDIM);
    pack_b_kernel<<<packGrid((size_t)1 * 4 * 4 * 32),    256, 0, stream>>>(fc1_w, fc1p, 1, 128, 64);

    int embTotal = BATCH * SEQ * FDIM;
    embed_kernel<<<(embTotal + 255) / 256, 256, 0, stream>>>(x, emb, actA, embTotal);

    dim3 pg(BATCH / 32 / 8, SEQ);   // 16 x 41 blocks, 8 waves x 32 rows each
    poslin_wmma<<<pg, 256, 0, stream>>>(actA, W3p, b3, actB, 384 / 32, 1);
    poslin_wmma<<<pg, 256, 0, stream>>>(actB, W5p, b5, actA, 640 / 32, 2);
    poslin_wmma<<<pg, 256, 0, stream>>>(actA, W7p, b7, actB, 896 / 32, 3);

    conv1_kernel<<<BATCH, 128, 0, stream>>>(actB, conv1_w, conv1_b, convout, stats);
    caps_route_kernel<<<BATCH, 128, 0, stream>>>(convout, stats, bn_g, bn_b,
                                                 caps_w, caps_b, route_W, route_b, ybuf);
    fc1_wmma<<<(BATCH * 2 / 16) / 8, 256, 0, stream>>>(ybuf, fc1p, z1, stats);
    fc2_kernel<<<BATCH * 2 / 8, 256, 0, stream>>>(z1, stats, bn1_g, bn1_b, fc2_w, z2);
    head_kernel<<<BATCH / 4, 256, 0, stream>>>(z2, stats, bn2_g, bn2_b, fc3_w, outp);
}
